// MultiHeadSelfAttention_86423331930199
// MI455X (gfx1250) — matrix operand order repaired and checked by reading, not compiled, not run on hardware
//
#include <hip/hip_runtime.h>

typedef __bf16 bf16;
typedef __attribute__((ext_vector_type(16))) __bf16 v16bf;
typedef __attribute__((ext_vector_type(8)))  float  v8f;
typedef __attribute__((ext_vector_type(4)))  unsigned int u32x4;
typedef __attribute__((ext_vector_type(8)))  unsigned int u32x8;

union FragB {
    v16bf v;
    uint4 q[2];
};

union Pack16 {
    bf16 h[16];
    uint4 q[2];
};

static __device__ inline v8f vzero() {
    v8f z = {0.f, 0.f, 0.f, 0.f, 0.f, 0.f, 0.f, 0.f};
    return z;
}

#define WMMA_BF16(A, B, C) \
    __builtin_amdgcn_wmma_f32_16x16x32_bf16(false, (A), false, (B), (short)0, (C), false, false)

// ---------------------------------------------------------------------------
// Kernel 1: QKV projection.  C[M=4096, E=1024] = X f32 * W^T -> bf16 in
// [B, H, N, DK].  Double-buffered LDS: one barrier per K-step, next tile's
// global loads issued before the current tile's WMMAs.
// ---------------------------------------------------------------------------
__global__ __launch_bounds__(256) void qkv_proj_kernel(
    const float* __restrict__ X, const float* __restrict__ W,
    const float* __restrict__ bias, bf16* __restrict__ out)
{
    __shared__ bf16 As[2][128][40];   // row stride 80B: 16B aligned, conflict-free
    __shared__ bf16 Bs[2][128][40];
    const int tid  = threadIdx.x;
    const int wave = tid >> 5, lane = tid & 31;
    const int half = lane >> 4, l15 = lane & 15;
    const int wrow = wave >> 1, wcol = wave & 1;      // 4x2 wave grid
    const int mtile = blockIdx.y * 128;
    const int ntile = blockIdx.x * 128;

    v8f acc[2][4];
    #pragma unroll
    for (int i = 0; i < 2; ++i)
        #pragma unroll
        for (int j = 0; j < 4; ++j) acc[i][j] = vzero();

    const int r  = tid >> 1;          // 0..127 : LDS row staged by this thread
    const int kc = (tid & 1) * 16;    // 0 or 16 : k-chunk

    auto loadCvt = [&](const float* base, int k0, Pack16& u) {
        const float4* p = (const float4*)(base + (size_t)k0 + kc);
        #pragma unroll
        for (int i = 0; i < 4; ++i) {
            float4 f = p[i];
            u.h[4*i+0] = (bf16)f.x; u.h[4*i+1] = (bf16)f.y;
            u.h[4*i+2] = (bf16)f.z; u.h[4*i+3] = (bf16)f.w;
        }
    };
    const float* xrow = X + (size_t)(mtile + r) * 1024;
    const float* wrowp = W + (size_t)(ntile + r) * 1024;

    Pack16 ua, ub;
    loadCvt(xrow, 0, ua);
    loadCvt(wrowp, 0, ub);
    *(uint4*)&As[0][r][kc] = ua.q[0];  *(uint4*)&As[0][r][kc + 8] = ua.q[1];
    *(uint4*)&Bs[0][r][kc] = ub.q[0];  *(uint4*)&Bs[0][r][kc + 8] = ub.q[1];
    __syncthreads();

    for (int k0 = 0; k0 < 1024; k0 += 32) {
        const int cur = (k0 >> 5) & 1;
        const bool more = (k0 + 32) < 1024;
        if (more) {                       // issue next tile's HBM loads early
            loadCvt(xrow, k0 + 32, ua);
            loadCvt(wrowp, k0 + 32, ub);
        }
        if (k0 + 64 < 1024) {             // keep HBM two steps ahead
            __builtin_prefetch(xrow + k0 + 64 + kc, 0, 3);
            __builtin_prefetch(wrowp + k0 + 64 + kc, 0, 3);
        }

        FragB a[2], wb[4];
        #pragma unroll
        for (int i = 0; i < 2; ++i) {   // A-frag: lane row = l15, ks 8h..8h+7, 16+8h..23+8h
            const bf16* p = &As[cur][wrow * 32 + i * 16 + l15][8 * half];
            a[i].q[0] = *(const uint4*)p;
            a[i].q[1] = *(const uint4*)(p + 16);
        }
        #pragma unroll
        for (int j = 0; j < 4; ++j) {   // B-frag: lane col = l15, ks 8h..8h+7, 16+8h..23+8h
            const bf16* p = &Bs[cur][wcol * 64 + j * 16 + l15][8 * half];
            wb[j].q[0] = *(const uint4*)p;
            wb[j].q[1] = *(const uint4*)(p + 16);
        }
        #pragma unroll
        for (int i = 0; i < 2; ++i)
            #pragma unroll
            for (int j = 0; j < 4; ++j)
                acc[i][j] = WMMA_BF16(a[i].v, wb[j].v, acc[i][j]);

        if (more) {                       // fill the other buffer
            const int nxt = cur ^ 1;
            *(uint4*)&As[nxt][r][kc] = ua.q[0];  *(uint4*)&As[nxt][r][kc + 8] = ua.q[1];
            *(uint4*)&Bs[nxt][r][kc] = ub.q[0];  *(uint4*)&Bs[nxt][r][kc + 8] = ub.q[1];
        }
        __syncthreads();
    }

    // epilogue: add bias, scatter to [B, H, N, DK] bf16
    #pragma unroll
    for (int j = 0; j < 4; ++j) {
        const int E  = ntile + wcol * 64 + j * 16 + l15;
        const float bv = bias[E];
        const int h = E >> 6, dk = E & 63;
        #pragma unroll
        for (int i = 0; i < 2; ++i) {
            #pragma unroll
            for (int rr = 0; rr < 8; ++rr) {
                const int M = mtile + wrow * 32 + i * 16 + rr + 8 * half;
                const int batch = M >> 10, n = M & 1023;
                out[(((size_t)batch * 16 + h) * 1024 + n) * 64 + dk] =
                    (bf16)(acc[i][j][rr] + bv);
            }
        }
    }
}

// ---------------------------------------------------------------------------
// Kernel 2: flash attention.  K tile staged by the Tensor Data Mover
// (tensor_load_to_lds, TENSORcnt) with hardware LDS row padding (144B rows);
// V tile transposed manually; P re-laid out through a per-wave LDS pad.
// ---------------------------------------------------------------------------
__global__ __launch_bounds__(256) void attn_kernel(
    const bf16* __restrict__ Q, const bf16* __restrict__ K,
    const bf16* __restrict__ V, const unsigned char* __restrict__ mask,
    bf16* __restrict__ O)
{
    __shared__ bf16 Ks[32][72];       // [key][d]  row stride 144B (TDM-padded)
    __shared__ bf16 Vs[64][40];       // [d][key]  transposed, stride 80B
    __shared__ bf16 Ps[8][16][40];    // per-wave P staging
    const int tid  = threadIdx.x;
    const int wave = tid >> 5, lane = tid & 31;
    const int half = lane >> 4, l15 = lane & 15;
    const int bh = blockIdx.y;
    const int b  = bh >> 4;
    const int qbase = blockIdx.x * 128 + wave * 16;
    const float scale = 0.125f;       // 1/sqrt(DK=64)
    const float NEG = -1e30f;

    FragB qa[2];                      // Q rows held in registers for whole pass
    #pragma unroll
    for (int t = 0; t < 2; ++t) {
        const bf16* p = Q + ((size_t)bh * 1024 + qbase + l15) * 64 + t * 32 + 8 * half;
        qa[t].q[0] = *(const uint4*)p;
        qa[t].q[1] = *(const uint4*)(p + 16);
    }

    float mrow[8], lrow[8];
    v8f o[4];
    #pragma unroll
    for (int rr = 0; rr < 8; ++rr) { mrow[rr] = NEG; lrow[rr] = 0.f; }
    #pragma unroll
    for (int t = 0; t < 4; ++t) o[t] = vzero();

    const int skey = tid >> 3;            // 0..31
    const int sd0  = (tid & 7) * 8;       // 0..56
    const unsigned int ksLds = (unsigned int)(size_t)&Ks[0][0];

    for (int ks = 0; ks < 1024; ks += 32) {
        __syncthreads();
        if (wave == 0) {
            // TDM: 2D tile, 32 rows x 64 bf16, row pad 16B -> 144B LDS rows.
            const unsigned long long gaddr =
                (unsigned long long)(const void*)(K + ((size_t)bh * 1024 + ks) * 64);
            u32x4 g0;
            g0[0] = 1u;                                         // count = 1 (valid D#)
            g0[1] = ksLds;                                      // lds_addr
            g0[2] = (unsigned int)gaddr;                        // global_addr[31:0]
            g0[3] = ((unsigned int)(gaddr >> 32) & 0x01FFFFFFu) // global_addr[56:32]
                    | 0x80000000u;                              // type = 2 ("image")
            u32x8 g1;
            g1[0] = (1u << 16)    // data_size = 2B
                  | (1u << 20)    // pad_enable
                  | (4u << 22)    // pad_interval: 32 DWORDs (one 128B row)
                  | (3u << 25);   // pad_amount: 4 DWORDs (16B)
            g1[1] = 64u << 16;    // tensor_dim0 = 64 (bits 79:48)
            g1[2] = 1024u << 16;  // tensor_dim1 = 1024 (bits 111:80)
            g1[3] = 64u << 16;    // tile_dim0 = 64 (bits 127:112)
            g1[4] = 32u;          // tile_dim1 = 32
            g1[5] = 64u;          // tensor_dim0_stride = 64 (bits 207:160)
            g1[6] = 0u;
            g1[7] = 0u;
            asm volatile("tensor_load_to_lds %0, %1" :: "s"(g0), "s"(g1) : "memory");
            __builtin_amdgcn_s_wait_tensorcnt(0);
        }
        {   // cooperative V staging with transpose (all 256 threads)
            const bf16* vsrc = V + ((size_t)bh * 1024 + ks + skey) * 64 + sd0;
            union { uint4 q; bf16 h[8]; } u;
            u.q = *(const uint4*)vsrc;
            #pragma unroll
            for (int j = 0; j < 8; ++j) Vs[sd0 + j][skey] = u.h[j];
            if (ks + 32 < 1024)
                __builtin_prefetch(vsrc + 32 * 64, 0, 3);
        }
        __syncthreads();

        // S = Q K^T : two 16x16 C tiles, K=64 via two bf16 WMMAs each
        v8f s[2];
        #pragma unroll
        for (int t = 0; t < 2; ++t) {
            v8f a = vzero();
            #pragma unroll
            for (int kk = 0; kk < 2; ++kk) {
                FragB kb;
                const bf16* p = &Ks[16 * t + l15][32 * kk + 8 * half];
                kb.q[0] = *(const uint4*)p;
                kb.q[1] = *(const uint4*)(p + 16);
                a = WMMA_BF16(qa[kk].v, kb.v, a);
            }
            s[t] = a;
        }

        const unsigned char msk0 = mask[(size_t)b * 1024 + ks + l15];
        const unsigned char msk1 = mask[(size_t)b * 1024 + ks + 16 + l15];

        #pragma unroll
        for (int rr = 0; rr < 8; ++rr) {
            float v0 = msk0 ? NEG : s[0][rr] * scale;
            float v1 = msk1 ? NEG : s[1][rr] * scale;
            float rm = fmaxf(v0, v1);
            #pragma unroll
            for (int d = 1; d < 16; d <<= 1) rm = fmaxf(rm, __shfl_xor(rm, d, 16));
            const float mnew  = fmaxf(mrow[rr], rm);
            const float alpha = __expf(mrow[rr] - mnew);
            mrow[rr] = mnew;
            const float p0 = __expf(v0 - mnew);
            const float p1 = __expf(v1 - mnew);
            s[0][rr] = p0; s[1][rr] = p1;
            float sum = p0 + p1;
            #pragma unroll
            for (int d = 1; d < 16; d <<= 1) sum += __shfl_xor(sum, d, 16);
            lrow[rr] = lrow[rr] * alpha + sum;
            #pragma unroll
            for (int t = 0; t < 4; ++t) o[t][rr] *= alpha;
        }

        // C-layout -> A-fragment relayout of P through per-wave LDS pad
        #pragma unroll
        for (int rr = 0; rr < 8; ++rr) {
            const int m = rr + 8 * half;
            Ps[wave][m][l15]      = (bf16)s[0][rr];
            Ps[wave][m][16 + l15] = (bf16)s[1][rr];
        }
        FragB pa;
        {
            const bf16* pp = &Ps[wave][l15][8 * half];
            pa.q[0] = *(const uint4*)pp;
            pa.q[1] = *(const uint4*)(pp + 16);
        }
        // O += P V : 4 WMMAs across DK=64
        #pragma unroll
        for (int t = 0; t < 4; ++t) {
            FragB vb;
            const bf16* vp = &Vs[16 * t + l15][8 * half];
            vb.q[0] = *(const uint4*)vp;
            vb.q[1] = *(const uint4*)(vp + 16);
            o[t] = WMMA_BF16(pa.v, vb.v, o[t]);
        }
    }

    float inv[8];
    #pragma unroll
    for (int rr = 0; rr < 8; ++rr) inv[rr] = lrow[rr] > 0.f ? 1.0f / lrow[rr] : 0.f;
    #pragma unroll
    for (int t = 0; t < 4; ++t) {
        #pragma unroll
        for (int rr = 0; rr < 8; ++rr) {
            const int q = qbase + rr + 8 * half;
            O[((size_t)bh * 1024 + q) * 64 + 16 * t + l15] = (bf16)(o[t][rr] * inv[rr]);
        }
    }
}

// ---------------------------------------------------------------------------
// Kernel 3: output projection.  A = attention output bf16 in [B,H,N,DK]
// (viewed as [4096 x 1024] head-interleaved); out = f32 [B,N,D].
// Same double-buffered pipeline as kernel 1.
// ---------------------------------------------------------------------------
__global__ __launch_bounds__(256) void out_proj_kernel(
    const bf16* __restrict__ A, const float* __restrict__ W,
    const float* __restrict__ bias, float* __restrict__ out)
{
    __shared__ bf16 As[2][128][40];
    __shared__ bf16 Bs[2][128][40];
    const int tid  = threadIdx.x;
    const int wave = tid >> 5, lane = tid & 31;
    const int half = lane >> 4, l15 = lane & 15;
    const int wrow = wave >> 1, wcol = wave & 1;
    const int mtile = blockIdx.y * 128;
    const int ntile = blockIdx.x * 128;

    v8f acc[2][4];
    #pragma unroll
    for (int i = 0; i < 2; ++i)
        #pragma unroll
        for (int j = 0; j < 4; ++j) acc[i][j] = vzero();

    const int r  = tid >> 1;
    const int kc = (tid & 1) * 16;
    const int M0 = mtile + r;
    const int batch = M0 >> 10, n0 = M0 & 1023;

    auto loadA = [&](int k0, Pack16& u) {
        const int c = k0 + kc;                 // 16-col run never crosses a head
        const int h = c >> 6, dk = c & 63;
        const uint4* ap =
            (const uint4*)(A + (((size_t)batch * 16 + h) * 1024 + n0) * 64 + dk);
        u.q[0] = ap[0]; u.q[1] = ap[1];
    };
    const float* wrowp = W + (size_t)(ntile + r) * 1024;
    auto loadW = [&](int k0, Pack16& u) {
        const float4* p = (const float4*)(wrowp + k0 + kc);
        #pragma unroll
        for (int i = 0; i < 4; ++i) {
            float4 f = p[i];
            u.h[4*i+0] = (bf16)f.x; u.h[4*i+1] = (bf16)f.y;
            u.h[4*i+2] = (bf16)f.z; u.h[4*i+3] = (bf16)f.w;
        }
    };

    Pack16 ua, ub;
    loadA(0, ua); loadW(0, ub);
    *(uint4*)&As[0][r][kc] = ua.q[0];  *(uint4*)&As[0][r][kc + 8] = ua.q[1];
    *(uint4*)&Bs[0][r][kc] = ub.q[0];  *(uint4*)&Bs[0][r][kc + 8] = ub.q[1];
    __syncthreads();

    for (int k0 = 0; k0 < 1024; k0 += 32) {
        const int cur = (k0 >> 5) & 1;
        const bool more = (k0 + 32) < 1024;
        if (more) { loadA(k0 + 32, ua); loadW(k0 + 32, ub); }
        if (k0 + 64 < 1024)
            __builtin_prefetch(wrowp + k0 + 64 + kc, 0, 3);

        FragB a[2], wb[4];
        #pragma unroll
        for (int i = 0; i < 2; ++i) {
            const bf16* p = &As[cur][wrow * 32 + i * 16 + l15][8 * half];
            a[i].q[0] = *(const uint4*)p;
            a[i].q[1] = *(const uint4*)(p + 16);
        }
        #pragma unroll
        for (int j = 0; j < 4; ++j) {
            const bf16* p = &Bs[cur][wcol * 64 + j * 16 + l15][8 * half];
            wb[j].q[0] = *(const uint4*)p;
            wb[j].q[1] = *(const uint4*)(p + 16);
        }
        #pragma unroll
        for (int i = 0; i < 2; ++i)
            #pragma unroll
            for (int j = 0; j < 4; ++j)
                acc[i][j] = WMMA_BF16(a[i].v, wb[j].v, acc[i][j]);

        if (more) {
            const int nxt = cur ^ 1;
            *(uint4*)&As[nxt][r][kc] = ua.q[0];  *(uint4*)&As[nxt][r][kc + 8] = ua.q[1];
            *(uint4*)&Bs[nxt][r][kc] = ub.q[0];  *(uint4*)&Bs[nxt][r][kc + 8] = ub.q[1];
        }
        __syncthreads();
    }

    #pragma unroll
    for (int j = 0; j < 4; ++j) {
        const int E  = ntile + wcol * 64 + j * 16 + l15;
        const float bv = bias[E];
        #pragma unroll
        for (int i = 0; i < 2; ++i) {
            #pragma unroll
            for (int rr = 0; rr < 8; ++rr) {
                const int M = mtile + wrow * 32 + i * 16 + rr + 8 * half;
                out[(size_t)M * 1024 + E] = acc[i][j][rr] + bv;   // [B,N,D] f32
            }
        }
    }
}

// ---------------------------------------------------------------------------
extern "C" void kernel_launch(void* const* d_in, const int* in_sizes, int n_in,
                              void* d_out, int out_size, void* d_ws, size_t ws_size,
                              hipStream_t stream) {
    const float* x          = (const float*)d_in[0];
    const unsigned char* mk = (const unsigned char*)d_in[1];
    const float* Wq = (const float*)d_in[2];
    const float* bq = (const float*)d_in[3];
    const float* Wk = (const float*)d_in[4];
    const float* bk = (const float*)d_in[5];
    const float* Wv = (const float*)d_in[6];
    const float* bv = (const float*)d_in[7];
    const float* Wo = (const float*)d_in[8];
    const float* bo = (const float*)d_in[9];

    const size_t per = (size_t)4 * 16 * 1024 * 64;   // 4,194,304 elems (8 MB bf16)
    bf16* Qb = (bf16*)d_ws;
    bf16* Kb = Qb + per;
    bf16* Vb = Kb + per;
    bf16* Ob = Vb + per;                              // total 32 MB of d_ws

    dim3 gemmGrid(8, 32);                             // 1024/128 x 4096/128
    qkv_proj_kernel<<<gemmGrid, 256, 0, stream>>>(x, Wq, bq, Qb);
    qkv_proj_kernel<<<gemmGrid, 256, 0, stream>>>(x, Wk, bk, Kb);
    qkv_proj_kernel<<<gemmGrid, 256, 0, stream>>>(x, Wv, bv, Vb);
    attn_kernel<<<dim3(8, 64), 256, 0, stream>>>(Qb, Kb, Vb, mk, Ob);
    out_proj_kernel<<<gemmGrid, 256, 0, stream>>>(Ob, Wo, bo, (float*)d_out);
}
